// MultiHeadAttention_3015067041874
// MI455X (gfx1250) — matrix operand order repaired and checked by reading, not compiled, not run on hardware
//
#include <hip/hip_runtime.h>
#include <hip/hip_bf16.h>

#define D_MODEL 1024
#define N_HEADS 16
#define HEAD_DIM 64
#define SEQ 2048
#define BATCH 2

typedef __bf16 bf16_t;
typedef __attribute__((ext_vector_type(16))) __bf16 v16bf;
typedef __attribute__((ext_vector_type(8)))  __bf16 v8bf;
typedef __attribute__((ext_vector_type(4)))  __bf16 v4bf;
typedef __attribute__((ext_vector_type(8)))  float  v8f;

static __device__ __forceinline__ v16bf combine8(v8bf lo, v8bf hi) {
    v16bf r;
#pragma unroll
    for (int i = 0; i < 8; i++) { r[i] = lo[i]; r[i + 8] = hi[i]; }
    return r;
}

static __device__ __forceinline__ v8f wmma_bf16(v16bf a, v16bf b, v8f c) {
    return __builtin_amdgcn_wmma_f32_16x16x32_bf16(false, a, false, b, (short)0, c,
                                                   false, false);
}

// Async global->LDS copy of 32 bytes (2 x b128), tracked by ASYNCcnt.
// Low 32 bits of a flat shared-memory pointer are the LDS byte offset
// (flat LDS aperture: addr[31:0] maps to LDS).
static __device__ __forceinline__ void async_copy32B(bf16_t* lds, const bf16_t* g) {
    unsigned l = (unsigned)(size_t)(void*)lds;
    asm volatile(
        "global_load_async_to_lds_b128 %0, %1, off\n\t"
        "global_load_async_to_lds_b128 %0, %1, off offset:16"
        :: "v"(l), "v"(g) : "memory");
}
static __device__ __forceinline__ void wait_async_le4() {
    asm volatile("s_wait_asynccnt 0x4" ::: "memory");
}
static __device__ __forceinline__ void wait_async_0() {
    asm volatile("s_wait_asynccnt 0x0" ::: "memory");
}

// ---------------------------------------------------------------------------
// fp32 -> bf16 cast, 4 elements per thread (row-major copy)
// ---------------------------------------------------------------------------
__global__ void cast4_kernel(const float* __restrict__ in, bf16_t* __restrict__ out, int n4) {
    int i = blockIdx.x * blockDim.x + threadIdx.x;
    if (i < n4) {
        float4 v = ((const float4*)in)[i];
        v4bf o;
        o[0] = (bf16_t)v.x; o[1] = (bf16_t)v.y; o[2] = (bf16_t)v.z; o[3] = (bf16_t)v.w;
        ((v4bf*)out)[i] = o;
    }
}

// fp32 [K][N] -> bf16 transposed [N][K] (one-time weight prep)
__global__ void castT_kernel(const float* __restrict__ in, bf16_t* __restrict__ out,
                             int K, int N) {
    int idx = blockIdx.x * blockDim.x + threadIdx.x;
    if (idx < K * N) {
        int k = idx / N, n = idx - k * N;
        out[(size_t)n * K + k] = (bf16_t)in[idx];
    }
}

// ---------------------------------------------------------------------------
// QKV projection: C[4096,3072] = Xbf[4096,1024] @ W (W given pre-transposed
// as Wt[3072][1024]).  Double-buffered async LDS staging, 8 waves, 128x128
// block tile, K-step 32.  Epilogue scatters Q,K -> [bh][T][Dh], V -> [bh][Dh][T].
// ---------------------------------------------------------------------------
__global__ __launch_bounds__(256)
void qkv_gemm_kernel(const bf16_t* __restrict__ A, const bf16_t* __restrict__ Bt,
                     const float* __restrict__ bias, bf16_t* __restrict__ Qd,
                     bf16_t* __restrict__ Kd, bf16_t* __restrict__ Vtd) {
    __shared__ __align__(32) bf16_t sA[2][128 * 32];
    __shared__ __align__(32) bf16_t sB[2][128 * 32];  // [n][k]
    const int tid = threadIdx.x, lane = tid & 31, wave = tid >> 5;
    const int wm = wave >> 2, wn = wave & 3;
    const int bm = blockIdx.y * 128, bn = blockIdx.x * 128;
    const int arow = tid >> 1, ahalf = (tid & 1) * 16;

    auto stage = [&](int it, int buf) {
        int kk = it * 32;
        async_copy32B(&sA[buf][arow * 32 + ahalf],
                      &A[(size_t)(bm + arow) * 1024 + kk + ahalf]);
        async_copy32B(&sB[buf][arow * 32 + ahalf],
                      &Bt[(size_t)(bn + arow) * 1024 + kk + ahalf]);
    };

    v8f acc[4][2];
    v8f zero8 = {0.f, 0.f, 0.f, 0.f, 0.f, 0.f, 0.f, 0.f};
#pragma unroll
    for (int i = 0; i < 4; i++)
#pragma unroll
        for (int j = 0; j < 2; j++) acc[i][j] = zero8;

    const int NK = 32;
    stage(0, 0);
    for (int it = 0; it < NK; it++) {
        const int buf = it & 1;
        if (it + 1 < NK) { stage(it + 1, buf ^ 1); wait_async_le4(); }
        else { wait_async_0(); }
        __syncthreads();

        const int r0 = (lane < 16) ? 0 : 8;     // A-fragment K run start
        const int koff = (lane < 16) ? 0 : 16;  // unused; B fragment also starts at r0
        v16bf af[4], bfr[2];
#pragma unroll
        for (int i = 0; i < 4; i++) {
            int row = wm * 64 + i * 16 + (lane & 15);
            v8bf lo = *(const v8bf*)&sA[buf][row * 32 + r0];
            v8bf hi = *(const v8bf*)&sA[buf][row * 32 + r0 + 16];
            af[i] = combine8(lo, hi);
        }
#pragma unroll
        for (int j = 0; j < 2; j++) {
            int n = wn * 32 + j * 16 + (lane & 15);
            bfr[j] = combine8(*(const v8bf*)&sB[buf][n * 32 + r0], *(const v8bf*)&sB[buf][n * 32 + r0 + 16]);
        }
#pragma unroll
        for (int i = 0; i < 4; i++)
#pragma unroll
            for (int j = 0; j < 2; j++) acc[i][j] = wmma_bf16(af[i], bfr[j], acc[i][j]);
        __syncthreads();
    }

    const int rbase = (lane < 16) ? 0 : 8;
#pragma unroll
    for (int i = 0; i < 4; i++)
#pragma unroll
        for (int j = 0; j < 2; j++)
#pragma unroll
            for (int r = 0; r < 8; r++) {
                int gr = bm + wm * 64 + i * 16 + r + rbase;
                int gc = bn + wn * 32 + j * 16 + (lane & 15);
                float v = acc[i][j][r] + bias[gc];
                int which = gc >> 10, d = gc & 1023;
                int h = d >> 6, dh = d & 63;
                int t = gr >> 1, bb = gr & 1;
                int bh = bb * 16 + h;
                if (which == 0)
                    Qd[((size_t)bh * SEQ + t) * HEAD_DIM + dh] = (bf16_t)v;
                else if (which == 1)
                    Kd[((size_t)bh * SEQ + t) * HEAD_DIM + dh] = (bf16_t)v;
                else
                    Vtd[((size_t)bh * HEAD_DIM + dh) * SEQ + t] = (bf16_t)v;
            }
}

// ---------------------------------------------------------------------------
// Flash attention: grid (32 qblocks, 32 b*h), 128 threads (4 waves).
// K in [bh][T][Dh], V pre-transposed [bh][Dh][T] -> all staging contiguous.
// Double-buffered async K/V tiles; WMMA bf16 for QK^T and PV.
// ---------------------------------------------------------------------------
__global__ __launch_bounds__(128)
void attn_kernel(const bf16_t* __restrict__ Qb, const bf16_t* __restrict__ Kb,
                 const bf16_t* __restrict__ Vtb, bf16_t* __restrict__ ctx) {
    __shared__ __align__(32) bf16_t sK[2][32 * 64];   // [key][dim]
    __shared__ __align__(32) bf16_t sVt[2][64 * 32];  // [dim][key]
    __shared__ __align__(32) bf16_t sP[4 * 16 * 40];  // per-wave P staging
    const int tid = threadIdx.x, lane = tid & 31, wave = tid >> 5;
    const int bh = blockIdx.y, b = bh >> 4, h = bh & 15;
    const int qblock = blockIdx.x;
    const int qbase = qblock * 64 + wave * 16;
    const bf16_t* Qp = Qb + (size_t)bh * SEQ * HEAD_DIM;
    const bf16_t* Kp = Kb + (size_t)bh * SEQ * HEAD_DIM;
    const bf16_t* Vtp = Vtb + (size_t)bh * HEAD_DIM * SEQ;

    const int skey = tid >> 2, sdoff = (tid & 3) * 16;  // K-tile staging coords
    const int sdim = tid >> 1, skoff = (tid & 1) * 16;  // Vt-tile staging coords
    auto stage = [&](int kb, int buf) {
        async_copy32B(&sK[buf][skey * 64 + sdoff],
                      &Kp[(size_t)(kb * 32 + skey) * 64 + sdoff]);
        async_copy32B(&sVt[buf][sdim * 32 + skoff],
                      &Vtp[(size_t)sdim * SEQ + kb * 32 + skoff]);
    };

    // Q fragments for the wave's 16 rows, two K-chunks of 32 dims
    v16bf aq[2];
    {
        int row = qbase + (lane & 15);
#pragma unroll
        for (int c = 0; c < 2; c++) {
            int o = c * 32 + ((lane < 16) ? 0 : 8);
            v8bf lo = *(const v8bf*)&Qp[(size_t)row * 64 + o];
            v8bf hi = *(const v8bf*)&Qp[(size_t)row * 64 + o + 16];
            aq[c] = combine8(lo, hi);
        }
    }

    float mrun[8], lrun[8];
    v8f acc[4];
    v8f zero8 = {0.f, 0.f, 0.f, 0.f, 0.f, 0.f, 0.f, 0.f};
#pragma unroll
    for (int r = 0; r < 8; r++) { mrun[r] = -3.0e38f; lrun[r] = 0.f; }
#pragma unroll
    for (int dt = 0; dt < 4; dt++) acc[dt] = zero8;

    const float sc = 0.125f * 1.44269504089f;  // 1/sqrt(64) * log2(e)
    const int nkb = qblock * 2 + 2;            // uniform trip count across block
    const int rb = (lane < 16) ? 0 : 8;
    const int koffB = (lane < 16) ? 0 : 16;

    stage(0, 0);
    for (int kb = 0; kb < nkb; kb++) {
        const int buf = kb & 1;
        if (kb + 1 < nkb) { stage(kb + 1, buf ^ 1); wait_async_le4(); }
        else { wait_async_0(); }
        __syncthreads();

        // S = Q K^T over 32 keys (two 16-key subtiles)
        v8f s0 = zero8, s1 = zero8;
#pragma unroll
        for (int c = 0; c < 2; c++) {
            v16bf b0 = combine8(*(const v8bf*)&sK[buf][(lane & 15) * 64 + c * 32 + rb], *(const v8bf*)&sK[buf][(lane & 15) * 64 + c * 32 + rb + 16]);
            s0 = wmma_bf16(aq[c], b0, s0);
            v16bf b1 = combine8(*(const v8bf*)&sK[buf][(16 + (lane & 15)) * 64 + c * 32 + rb], *(const v8bf*)&sK[buf][(16 + (lane & 15)) * 64 + c * 32 + rb + 16]);
            s1 = wmma_bf16(aq[c], b1, s1);
        }

        // scale + causal mask
        const int kcol = kb * 32 + (lane & 15);
#pragma unroll
        for (int r = 0; r < 8; r++) {
            int q = qbase + r + rb;
            s0[r] = (kcol <= q) ? s0[r] * sc : -3.0e38f;
            s1[r] = (kcol + 16 <= q) ? s1[r] * sc : -3.0e38f;
        }

        // online softmax update (base-2 domain)
#pragma unroll
        for (int r = 0; r < 8; r++) {
            float mt = fmaxf(s0[r], s1[r]);
            for (int o = 1; o < 16; o <<= 1) mt = fmaxf(mt, __shfl_xor(mt, o, 32));
            float mnew = fmaxf(mrun[r], mt);
            float alpha = exp2f(mrun[r] - mnew);
            mrun[r] = mnew;
            float p0 = exp2f(s0[r] - mnew);
            float p1 = exp2f(s1[r] - mnew);
            s0[r] = p0; s1[r] = p1;
            float rs = p0 + p1;
            for (int o = 1; o < 16; o <<= 1) rs += __shfl_xor(rs, o, 32);
            lrun[r] = lrun[r] * alpha + rs;
#pragma unroll
            for (int dt = 0; dt < 4; dt++) acc[dt][r] *= alpha;
        }

        // stage P (16x32) to wave-private LDS, reload as A-fragment
        bf16_t* pw = &sP[wave * 16 * 40];
        {
            int col = lane & 15;
#pragma unroll
            for (int r = 0; r < 8; r++) {
                int row = r + rb;
                pw[row * 40 + col] = (bf16_t)s0[r];
                pw[row * 40 + 16 + col] = (bf16_t)s1[r];
            }
        }
        asm volatile("s_wait_dscnt 0" ::: "memory");
        v16bf pa;
        {
            int row = lane & 15;
            int o = (lane < 16) ? 0 : 8;
            v8bf lo = *(const v8bf*)&pw[row * 40 + o];
            v8bf hi = *(const v8bf*)&pw[row * 40 + o + 16];
            pa = combine8(lo, hi);
        }
        // context += P @ V  (4 dim tiles of 16)
#pragma unroll
        for (int dt = 0; dt < 4; dt++) {
            v16bf bv = combine8(*(const v8bf*)&sVt[buf][(dt * 16 + (lane & 15)) * 32 + rb], *(const v8bf*)&sVt[buf][(dt * 16 + (lane & 15)) * 32 + rb + 16]);
            acc[dt] = wmma_bf16(pa, bv, acc[dt]);
        }
        __syncthreads();
    }

    // normalize and store context into [T,B,D] bf16
#pragma unroll
    for (int r = 0; r < 8; r++) {
        float inv = 1.0f / lrun[r];
        int tq = qbase + r + rb;
        size_t rowoff = ((size_t)tq * BATCH + b) * D_MODEL + h * 64;
#pragma unroll
        for (int dt = 0; dt < 4; dt++)
            ctx[rowoff + dt * 16 + (lane & 15)] = (bf16_t)(acc[dt][r] * inv);
    }
}

// ---------------------------------------------------------------------------
// Output projection: out[4096,1024] = ctx_bf16 @ Wout (pre-transposed
// Wt[1024][1024]) + b_out, fp32 output.  Same double-buffered async scheme.
// ---------------------------------------------------------------------------
__global__ __launch_bounds__(256)
void out_gemm_kernel(const bf16_t* __restrict__ A, const bf16_t* __restrict__ Bt,
                     const float* __restrict__ bias, float* __restrict__ out) {
    __shared__ __align__(32) bf16_t sA[2][128 * 32];
    __shared__ __align__(32) bf16_t sB[2][128 * 32];
    const int tid = threadIdx.x, lane = tid & 31, wave = tid >> 5;
    const int wm = wave >> 2, wn = wave & 3;
    const int bm = blockIdx.y * 128, bn = blockIdx.x * 128;
    const int arow = tid >> 1, ahalf = (tid & 1) * 16;

    auto stage = [&](int it, int buf) {
        int kk = it * 32;
        async_copy32B(&sA[buf][arow * 32 + ahalf],
                      &A[(size_t)(bm + arow) * 1024 + kk + ahalf]);
        async_copy32B(&sB[buf][arow * 32 + ahalf],
                      &Bt[(size_t)(bn + arow) * 1024 + kk + ahalf]);
    };

    v8f acc[4][2];
    v8f zero8 = {0.f, 0.f, 0.f, 0.f, 0.f, 0.f, 0.f, 0.f};
#pragma unroll
    for (int i = 0; i < 4; i++)
#pragma unroll
        for (int j = 0; j < 2; j++) acc[i][j] = zero8;

    const int NK = 32;
    stage(0, 0);
    for (int it = 0; it < NK; it++) {
        const int buf = it & 1;
        if (it + 1 < NK) { stage(it + 1, buf ^ 1); wait_async_le4(); }
        else { wait_async_0(); }
        __syncthreads();

        const int r0 = (lane < 16) ? 0 : 8;
        const int koff = (lane < 16) ? 0 : 16;
        v16bf af[4], bfr[2];
#pragma unroll
        for (int i = 0; i < 4; i++) {
            int row = wm * 64 + i * 16 + (lane & 15);
            v8bf lo = *(const v8bf*)&sA[buf][row * 32 + r0];
            v8bf hi = *(const v8bf*)&sA[buf][row * 32 + r0 + 16];
            af[i] = combine8(lo, hi);
        }
#pragma unroll
        for (int j = 0; j < 2; j++) {
            int n = wn * 32 + j * 16 + (lane & 15);
            bfr[j] = combine8(*(const v8bf*)&sB[buf][n * 32 + r0], *(const v8bf*)&sB[buf][n * 32 + r0 + 16]);
        }
#pragma unroll
        for (int i = 0; i < 4; i++)
#pragma unroll
            for (int j = 0; j < 2; j++) acc[i][j] = wmma_bf16(af[i], bfr[j], acc[i][j]);
        __syncthreads();
    }

    const int rbase = (lane < 16) ? 0 : 8;
#pragma unroll
    for (int i = 0; i < 4; i++)
#pragma unroll
        for (int j = 0; j < 2; j++)
#pragma unroll
            for (int r = 0; r < 8; r++) {
                int gr = bm + wm * 64 + i * 16 + r + rbase;
                int gc = bn + wn * 32 + j * 16 + (lane & 15);
                out[(size_t)gr * 1024 + gc] = acc[i][j][r] + bias[gc];
            }
}

// ---------------------------------------------------------------------------
extern "C" void kernel_launch(void* const* d_in, const int* in_sizes, int n_in,
                              void* d_out, int out_size, void* d_ws, size_t ws_size,
                              hipStream_t stream) {
    const float* x = (const float*)d_in[0];
    const float* Wqkv = (const float*)d_in[1];
    const float* bqkv = (const float*)d_in[2];
    const float* Wout = (const float*)d_in[3];
    const float* bout = (const float*)d_in[4];

    char* ws = (char*)d_ws;
    bf16_t* xb  = (bf16_t*)(ws);                  // 8 MB: x bf16 [4096,1024]
    bf16_t* wqt = (bf16_t*)(ws + (8ull << 20));   // 6 MB: Wqkv^T bf16 [3072,1024]
    bf16_t* wot = (bf16_t*)(ws + (14ull << 20));  // 2 MB: Wout^T bf16 [1024,1024]
    bf16_t* Qd  = (bf16_t*)(ws + (16ull << 20));  // 8 MB: Q [32][2048][64]
    bf16_t* Kd  = (bf16_t*)(ws + (24ull << 20));  // 8 MB: K [32][2048][64]
    bf16_t* Vtd = (bf16_t*)(ws + (32ull << 20));  // 8 MB: V^T [32][64][2048]
    bf16_t* ctxb = (bf16_t*)(ws + (40ull << 20)); // 8 MB: ctx [4096,1024]

    const int nx = 4096 * 1024, nwq = 1024 * 3072, nwo = 1024 * 1024;
    cast4_kernel<<<nx / 4 / 256, 256, 0, stream>>>(x, xb, nx / 4);
    castT_kernel<<<nwq / 256, 256, 0, stream>>>(Wqkv, wqt, 1024, 3072);
    castT_kernel<<<nwo / 256, 256, 0, stream>>>(Wout, wot, 1024, 1024);

    qkv_gemm_kernel<<<dim3(24, 32), 256, 0, stream>>>(xb, wqt, bqkv, Qd, Kd, Vtd);

    attn_kernel<<<dim3(32, 32), 128, 0, stream>>>(Qd, Kd, Vtd, ctxb);

    out_gemm_kernel<<<dim3(8, 32), 256, 0, stream>>>(ctxb, wot, bout, (float*)d_out);
}
